// RGNN_44693429682201
// MI455X (gfx1250) — compile-verified, matrix operand order checked by reading, not run on hardware
//
#include <hip/hip_runtime.h>
#include <hip/hip_bf16.h>

// ---------------------------------------------------------------------------
// RGNN forward for MI455X (gfx1250, wave32).
// Dense GEMMs (x@Wl / xt@Wr / skip / proj) run on v_wmma_f32_16x16x32_bf16
// with fp32 accumulation; relation weights are fused into one N=512 GEMM.
// Tile staging uses the CDNA5 data movers:
//   - A tile: global_load_async_to_lds_b128 (ASYNCcnt)
//   - B tile: TDM tensor_load_to_lds with an SGPR D# descriptor (TENSORcnt)
// Edge softmax uses encoded-uint atomicMax + f32 atomicAdd (L2 atomics).
// ---------------------------------------------------------------------------

typedef __attribute__((ext_vector_type(16))) __bf16 v16bf;
typedef __attribute__((ext_vector_type(8)))  float  v8f;
typedef __attribute__((ext_vector_type(4)))  unsigned int u32x4;
typedef __attribute__((ext_vector_type(8)))  unsigned int u32x8;

union WFrag {            // 16 bf16 = 32 bytes = 8 VGPRs (wave32 WMMA operand)
    v16bf v;
    uint4 q[2];
};

#define KDIM 128
#define BM   128
#define BN   64

__device__ inline unsigned short f2bf(float f) {
    unsigned int u = __float_as_uint(f);
    unsigned int r = u + 0x7FFFu + ((u >> 16) & 1u);   // round-to-nearest-even
    return (unsigned short)(r >> 16);
}
// order-preserving float<->uint encoding for atomicMax on floats
__device__ inline unsigned int fenc(float f) {
    unsigned int u = __float_as_uint(f);
    return (u & 0x80000000u) ? ~u : (u | 0x80000000u);
}
__device__ inline float fdec(unsigned int k) {
    unsigned int u = (k & 0x80000000u) ? (k & 0x7FFFFFFFu) : ~k;
    return __uint_as_float(u);
}

// LDS flat addresses keep the wave-relative byte offset in addr[31:0]
// (ISA 10.2 aperture mapping), so truncation yields the DS/TDM LDS address.
__device__ inline unsigned lds_off(const void* p) {
    return (unsigned)(size_t)p;
}

__device__ inline void async_g2l_b128(unsigned ldsaddr, const void* gaddr) {
    unsigned long long ga = (unsigned long long)(size_t)gaddr;
    asm volatile("global_load_async_to_lds_b128 %0, %1, off"
                 :: "v"(ldsaddr), "v"(ga) : "memory");
}

__device__ inline void wait_async0() {
#if __has_builtin(__builtin_amdgcn_s_wait_asynccnt)
    __builtin_amdgcn_s_wait_asynccnt(0);
#else
    asm volatile("s_wait_asynccnt 0" ::: "memory");
#endif
}

// ---------------------------------------------------------------------------
// WMMA GEMM: C[M x N] = A[M x 128](bf16) * B[128 x N](bf16, given transposed
// as Bt[N x 128]) + bias[N].  Block tile 128x64, 8 waves, full K in LDS.
// A-fragment layout (16-bit A 16x32, ISA 7.12.2): lanes 0-15 row M=lane,
// halves K={k0..k0+7, k0+16..k0+23}; lanes 16-31 same row, K={+8, +24}.
// Bt rows are columns of B, so B fragments load with the identical pattern.
// ---------------------------------------------------------------------------
__global__ __launch_bounds__(256) void rgnn_gemm_bf16(
    const unsigned short* __restrict__ A,
    const unsigned short* __restrict__ Bt,
    const float* __restrict__ bias,
    float* __restrict__ C,
    int M, int N)
{
    __shared__ __align__(16) unsigned short sA[BM * KDIM];
    __shared__ __align__(16) unsigned short sB[BN * KDIM];

    const int t  = threadIdx.x;
    const int bm = blockIdx.x * BM;
    const int bn = blockIdx.y * BN;

    if (t < BN) __builtin_prefetch(bias + bn + t, 0, 0);  // global_prefetch_b8

    {   // A tile: 2 threads/row, 64 cols each; async direct-to-LDS, zero-pad ragged M
        int row = t >> 1, half = t & 1;
        unsigned short* ld = &sA[row * KDIM + half * 64];
        if (bm + row < M) {
            const unsigned short* g = A + (size_t)(bm + row) * KDIM + half * 64;
#pragma unroll
            for (int q = 0; q < 8; ++q)
                async_g2l_b128(lds_off(ld + q * 8), g + q * 8);
        } else {
            uint4 z = make_uint4(0u, 0u, 0u, 0u);
            uint4* d = (uint4*)ld;
#pragma unroll
            for (int q = 0; q < 8; ++q) d[q] = z;
        }
    }

    // B tile: one TDM DMA for the whole 64x128 bf16 tile, issued by wave 0.
    // D# group0: count=1 | lds_addr | global_addr(57b) | type=2.
    // D# group1: data_size=1(2B), tensor_dim0=128, tensor_dim1=64 rows,
    //            tile_dim0=128, tile_dim1=64, tensor_dim0_stride=128.
    if ((t >> 5) == 0) {
        unsigned long long gB = (unsigned long long)(size_t)(const void*)
                                (Bt + (size_t)bn * KDIM);
        u32x4 g0;
        g0[0] = 1u;                                     // count=1 (user descriptor)
        g0[1] = lds_off(&sB[0]);                        // lds_addr (bytes)
        g0[2] = (unsigned)(gB & 0xFFFFFFFFull);         // global_addr[31:0]
        g0[3] = (unsigned)((gB >> 32) & 0x01FFFFFFull)  // global_addr[56:32]
              | (2u << 30);                             // type=2 ("image")
        u32x8 g1;
        g1[0] = 0x00010000u;          // workgroup_mask=0, data_size=1 (2 bytes)
        g1[1] = 128u << 16;           // tensor_dim0[15:0] at bits 63:48
        g1[2] = (unsigned)BN << 16;   // tensor_dim1[15:0] at bits 95:80
        g1[3] = 128u << 16;           // tile_dim0 at bits 127:112
        g1[4] = (unsigned)BN;         // tile_dim1 at bits 143:128 (tile_dim2=0)
        g1[5] = 128u;                 // tensor_dim0_stride[31:0]
        g1[6] = 0u;
        g1[7] = 0u;
        asm volatile("tensor_load_to_lds %0, %1" :: "s"(g0), "s"(g1) : "memory");
        __builtin_amdgcn_s_wait_tensorcnt(0);
    }

    wait_async0();
    __syncthreads();

    const int lane = t & 31;
    const int wid  = t >> 5;
    const int wm   = (wid & 3) * 32;   // wave row offset within block tile
    const int wn   = (wid >> 2) * 32;  // wave col offset within block tile
    const int lrow = lane & 15;
    const int hsel = lane >> 4;

    v8f acc[2][2];
#pragma unroll
    for (int i = 0; i < 2; ++i)
#pragma unroll
        for (int j = 0; j < 2; ++j)
            acc[i][j] = (v8f){0.f, 0.f, 0.f, 0.f, 0.f, 0.f, 0.f, 0.f};

#pragma unroll
    for (int kb = 0; kb < 4; ++kb) {
        const int klo = kb * 32 + hsel * 8;
        WFrag a0, a1, b0, b1;
        {
            const unsigned short* p = &sA[(wm + lrow) * KDIM + klo];
            a0.q[0] = *(const uint4*)(p);
            a0.q[1] = *(const uint4*)(p + 16);
        }
        {
            const unsigned short* p = &sA[(wm + 16 + lrow) * KDIM + klo];
            a1.q[0] = *(const uint4*)(p);
            a1.q[1] = *(const uint4*)(p + 16);
        }
        {
            const unsigned short* p = &sB[(wn + lrow) * KDIM + klo];
            b0.q[0] = *(const uint4*)(p);
            b0.q[1] = *(const uint4*)(p + 16);
        }
        {
            const unsigned short* p = &sB[(wn + 16 + lrow) * KDIM + klo];
            b1.q[0] = *(const uint4*)(p);
            b1.q[1] = *(const uint4*)(p + 16);
        }
        acc[0][0] = __builtin_amdgcn_wmma_f32_16x16x32_bf16(false, a0.v, false, b0.v,
                                                            (short)0, acc[0][0], false, false);
        acc[0][1] = __builtin_amdgcn_wmma_f32_16x16x32_bf16(false, a0.v, false, b1.v,
                                                            (short)0, acc[0][1], false, false);
        acc[1][0] = __builtin_amdgcn_wmma_f32_16x16x32_bf16(false, a1.v, false, b0.v,
                                                            (short)0, acc[1][0], false, false);
        acc[1][1] = __builtin_amdgcn_wmma_f32_16x16x32_bf16(false, a1.v, false, b1.v,
                                                            (short)0, acc[1][1], false, false);
    }

    // C/D layout (32-bit 16x16): VGPR r, lanes 0-15: (M=r, N=lane);
    // lanes 16-31: (M=r+8, N=lane-16)
#pragma unroll
    for (int ms = 0; ms < 2; ++ms) {
#pragma unroll
        for (int ns = 0; ns < 2; ++ns) {
            const int gcol = bn + wn + ns * 16 + (lane & 15);
            const float bv = bias[gcol];
#pragma unroll
            for (int r = 0; r < 8; ++r) {
                const int grow = bm + wm + ms * 16 + (lane >> 4) * 8 + r;
                if (grow < M) C[(size_t)grow * N + gcol] = acc[ms][ns][r] + bv;
            }
        }
    }
}

// ---------------------------------------------------------------------------
// Small helper kernels
// ---------------------------------------------------------------------------
__global__ __launch_bounds__(256) void rgnn_f2bf(const float* __restrict__ x,
                                                 unsigned short* __restrict__ y, int n) {
    int t = blockIdx.x * 256 + threadIdx.x;
    if (t < n) y[t] = f2bf(x[t]);
}

// transpose a 128 x ncols fp32 block (row stride ldn) into bf16 [ncols x 128]
__global__ __launch_bounds__(256) void rgnn_transpose_bf16(
    const float* __restrict__ W, int ldn, unsigned short* __restrict__ Wt, int ncols) {
    int t = blockIdx.x * 256 + threadIdx.x;
    if (t >= 128 * ncols) return;
    int k = t / ncols, n = t % ncols;
    Wt[(size_t)n * 128 + k] = f2bf(W[(size_t)k * ldn + n]);
}

__global__ __launch_bounds__(256) void rgnn_init_layer(
    float* __restrict__ agg, unsigned int* __restrict__ amax,
    float* __restrict__ den, int nt) {
    int t = blockIdx.x * 256 + threadIdx.x;
    if (t < nt * 512) agg[t] = 0.f;
    if (t < nt * 16) { amax[t] = fenc(-1.0e9f); den[t] = 0.f; }
}

// pass 1: per (edge, head) attention logit + segment max over (dst, rel, head)
__global__ __launch_bounds__(256) void rgnn_edge_logit(
    const int* __restrict__ src, const int* __restrict__ dst, const int* __restrict__ ea,
    const float* __restrict__ xl, const float* __restrict__ xr,
    const float* __restrict__ att,              // layer slice [R][4][32]
    float* __restrict__ elog, unsigned int* __restrict__ amax, int E) {
    long t = (long)blockIdx.x * 256 + threadIdx.x;
    if (t >= (long)E * 4) return;
    int e = (int)(t >> 2), h = (int)(t & 3);
    int j = ea[e], s = src[e], d = dst[e];
    const float4* xl4 = (const float4*)(xl + (size_t)s * 512 + (size_t)j * 128 + h * 32);
    const float4* xr4 = (const float4*)(xr + (size_t)d * 512 + (size_t)j * 128 + h * 32);
    const float4* at4 = (const float4*)(att + ((size_t)j * 4 + h) * 32);
    float acc = 0.f;
#pragma unroll
    for (int q = 0; q < 8; ++q) {
        float4 a = xl4[q], b = xr4[q], w = at4[q];
        float m;
        m = a.x + b.x; acc += (m > 0.f ? m : 0.2f * m) * w.x;
        m = a.y + b.y; acc += (m > 0.f ? m : 0.2f * m) * w.y;
        m = a.z + b.z; acc += (m > 0.f ? m : 0.2f * m) * w.z;
        m = a.w + b.w; acc += (m > 0.f ? m : 0.2f * m) * w.w;
    }
    elog[t] = acc;
    atomicMax(&amax[(size_t)d * 16 + j * 4 + h], fenc(acc));
}

// pass 2: p = exp(logit - amax), denominator accumulation
__global__ __launch_bounds__(256) void rgnn_edge_p(
    const int* __restrict__ dst, const int* __restrict__ ea,
    float* __restrict__ elog, const unsigned int* __restrict__ amax,
    float* __restrict__ den, int E) {
    long t = (long)blockIdx.x * 256 + threadIdx.x;
    if (t >= (long)E * 4) return;
    int e = (int)(t >> 2), h = (int)(t & 3);
    int j = ea[e], d = dst[e];
    float am = fdec(amax[(size_t)d * 16 + j * 4 + h]);
    float p = __expf(elog[t] - am);
    elog[t] = p;
    atomicAdd(&den[(size_t)d * 16 + j * 4 + h], p);
}

// pass 3: agg[dst, rel, :] += alpha * xl[src, rel, :]
__global__ __launch_bounds__(256) void rgnn_edge_agg(
    const int* __restrict__ src, const int* __restrict__ dst, const int* __restrict__ ea,
    const float* __restrict__ xl, const float* __restrict__ elog,
    const float* __restrict__ den, float* __restrict__ agg, int E) {
    long t = (long)blockIdx.x * 256 + threadIdx.x;
    if (t >= (long)E * 128) return;
    int e = (int)(t >> 7), k = (int)(t & 127), h = k >> 5;
    int j = ea[e], s = src[e], d = dst[e];
    float al = elog[(size_t)e * 4 + h] / fmaxf(den[(size_t)d * 16 + j * 4 + h], 1e-16f);
    atomicAdd(&agg[(size_t)d * 512 + (size_t)j * 128 + k],
              al * xl[(size_t)s * 512 + (size_t)j * 128 + k]);
}

// node epilogue: skip + sum_j BN(agg_j + conv_b_j), then ELU; emit bf16 (+f32)
__global__ __launch_bounds__(256) void rgnn_node_final(
    const float* __restrict__ skipC, const float* __restrict__ agg,
    const float* __restrict__ conv_b, const float* __restrict__ bn_g,
    const float* __restrict__ bn_b, const float* __restrict__ bn_m,
    const float* __restrict__ bn_v,
    unsigned short* __restrict__ xbf, float* __restrict__ xout, int nt) {
    int t = blockIdx.x * 256 + threadIdx.x;
    if (t >= nt * 128) return;
    int n = t >> 7, k = t & 127;
    float o = skipC[t];
#pragma unroll
    for (int j = 0; j < 4; ++j) {
        int pi = j * 128 + k;
        float a = agg[(size_t)n * 512 + j * 128 + k] + conv_b[pi];
        o += (a - bn_m[pi]) * rsqrtf(bn_v[pi] + 1e-5f) * bn_g[pi] + bn_b[pi];
    }
    float e = o > 0.f ? o : expm1f(o);
    xbf[t] = f2bf(e);
    if (xout) xout[t] = e;
}

__global__ __launch_bounds__(256) void rgnn_proj_final(
    const float* __restrict__ h, const float* __restrict__ g, const float* __restrict__ b,
    const float* __restrict__ m, const float* __restrict__ v,
    const float* __restrict__ a, float* __restrict__ out, int n) {
    int t = blockIdx.x * 256 + threadIdx.x;
    if (t >= n) return;
    int k = t & 127;
    float hv = h[t];
    float hb = (hv - m[k]) * rsqrtf(v[k] + 1e-5f) * g[k] + b[k];
    out[t] = hb > 0.f ? hb : a[0] * hb;
}

// ---------------------------------------------------------------------------
// Host orchestration
// ---------------------------------------------------------------------------
static inline dim3 g1(long n) { return dim3((unsigned)((n + 255) / 256)); }

extern "C" void kernel_launch(void* const* d_in, const int* in_sizes, int n_in,
                              void* d_out, int out_size, void* d_ws, size_t ws_size,
                              hipStream_t stream) {
    const float* x0     = (const float*)d_in[0];
    const int*   src0   = (const int*)d_in[1];
    const int*   dst0   = (const int*)d_in[2];
    const int*   ea0    = (const int*)d_in[3];
    const int*   src1   = (const int*)d_in[4];
    const int*   dst1   = (const int*)d_in[5];
    const int*   ea1    = (const int*)d_in[6];
    const float* Wl     = (const float*)d_in[7];
    const float* Wr     = (const float*)d_in[8];
    const float* bl     = (const float*)d_in[9];
    const float* br     = (const float*)d_in[10];
    const float* att    = (const float*)d_in[11];
    const float* conv_b = (const float*)d_in[12];
    const float* bn_g   = (const float*)d_in[13];
    const float* bn_b   = (const float*)d_in[14];
    const float* bn_m   = (const float*)d_in[15];
    const float* bn_v   = (const float*)d_in[16];
    const float* skip_W = (const float*)d_in[17];
    const float* skip_b = (const float*)d_in[18];
    const float* proj_W = (const float*)d_in[19];
    const float* proj_b = (const float*)d_in[20];
    const float* pbn_g  = (const float*)d_in[21];
    const float* pbn_b  = (const float*)d_in[22];
    const float* pbn_m  = (const float*)d_in[23];
    const float* pbn_v  = (const float*)d_in[24];
    const float* prelua = (const float*)d_in[25];

    const int N0  = in_sizes[0] / 128;   // 100000
    const int E0  = in_sizes[1];         // 400000
    const int E1  = in_sizes[4];         // 150000
    const int NT0 = 30000;               // n_t0 (static in setup_inputs)
    const int NT1 = 8000;                // n_t1

    // workspace carve-out (256B-aligned)
    size_t off = 0;
    char* wsb = (char*)d_ws;
    auto carve = [&](size_t bytes) -> char* {
        char* p = wsb + off;
        off += (bytes + 255) & ~(size_t)255;
        return p;
    };
    float*          XL   = (float*)carve((size_t)N0  * 512 * 4);
    float*          XR   = (float*)carve((size_t)NT0 * 512 * 4);
    float*          AGG  = (float*)carve((size_t)NT0 * 512 * 4);
    float*          SKIP = (float*)carve((size_t)NT0 * 128 * 4);
    unsigned short* XBF  = (unsigned short*)carve((size_t)N0 * 128 * 2);
    float*          ELOG = (float*)carve((size_t)E0 * 4 * 4);
    unsigned int*   AMAX = (unsigned int*)carve((size_t)NT0 * 16 * 4);
    float*          DEN  = (float*)carve((size_t)NT0 * 16 * 4);
    unsigned short* WLT  = (unsigned short*)carve((size_t)2 * 512 * 128 * 2);
    unsigned short* WRT  = (unsigned short*)carve((size_t)2 * 512 * 128 * 2);
    unsigned short* SKT  = (unsigned short*)carve((size_t)2 * 128 * 128 * 2);
    unsigned short* PRT  = (unsigned short*)carve((size_t)128 * 128 * 2);
    float*          PH   = (float*)carve((size_t)NT1 * 128 * 4);

    // weights -> bf16, transposed to [N x K] for the WMMA B-operand
    for (int i = 0; i < 2; ++i) {
        for (int j = 0; j < 4; ++j) {
            rgnn_transpose_bf16<<<g1(128 * 128), 256, 0, stream>>>(
                Wl + (size_t)(i * 4 + j) * 128 * 128, 128,
                WLT + (size_t)i * 512 * 128 + (size_t)j * 128 * 128, 128);
            rgnn_transpose_bf16<<<g1(128 * 128), 256, 0, stream>>>(
                Wr + (size_t)(i * 4 + j) * 128 * 128, 128,
                WRT + (size_t)i * 512 * 128 + (size_t)j * 128 * 128, 128);
        }
        rgnn_transpose_bf16<<<g1(128 * 128), 256, 0, stream>>>(
            skip_W + (size_t)i * 128 * 128, 128, SKT + (size_t)i * 128 * 128, 128);
    }
    rgnn_transpose_bf16<<<g1(128 * 128), 256, 0, stream>>>(proj_W, 128, PRT, 128);

    rgnn_f2bf<<<g1((long)N0 * 128), 256, 0, stream>>>(x0, XBF, N0 * 128);

    auto gemm = [&](const unsigned short* A, const unsigned short* Bt,
                    const float* bias, float* C, int M, int N) {
        dim3 grid((M + BM - 1) / BM, N / BN);
        rgnn_gemm_bf16<<<grid, 256, 0, stream>>>(A, Bt, bias, C, M, N);
    };

    auto run_layer = [&](int li, const int* src, const int* dst, const int* ea,
                         int E, int Ms, int nt, float* xout) {
        gemm(XBF, WLT + (size_t)li * 512 * 128, bl + li * 512, XL, Ms, 512);
        gemm(XBF, WRT + (size_t)li * 512 * 128, br + li * 512, XR, nt, 512);
        gemm(XBF, SKT + (size_t)li * 128 * 128, skip_b + li * 128, SKIP, nt, 128);
        rgnn_init_layer<<<g1((long)nt * 512), 256, 0, stream>>>(AGG, AMAX, DEN, nt);
        rgnn_edge_logit<<<g1((long)E * 4), 256, 0, stream>>>(
            src, dst, ea, XL, XR, att + li * 512, ELOG, AMAX, E);
        rgnn_edge_p<<<g1((long)E * 4), 256, 0, stream>>>(dst, ea, ELOG, AMAX, DEN, E);
        rgnn_edge_agg<<<g1((long)E * 128), 256, 0, stream>>>(
            src, dst, ea, XL, ELOG, DEN, AGG, E);
        rgnn_node_final<<<g1((long)nt * 128), 256, 0, stream>>>(
            SKIP, AGG, conv_b + li * 512, bn_g + li * 512, bn_b + li * 512,
            bn_m + li * 512, bn_v + li * 512, XBF, xout, nt);
    };

    run_layer(0, src0, dst0, ea0, E0, N0, NT0, nullptr);
    run_layer(1, src1, dst1, ea1, E1, NT0, NT1, (float*)d_out);  // x -> out[0:NT1*128]

    gemm(XBF, PRT, proj_b, PH, NT1, 128);
    rgnn_proj_final<<<g1((long)NT1 * 128), 256, 0, stream>>>(
        PH, pbn_g, pbn_b, pbn_m, pbn_v, prelua,
        (float*)d_out + (size_t)NT1 * 128, NT1 * 128);
}
